// CenterHeadSingle_88819923681855
// MI455X (gfx1250) — matrix operand order repaired and checked by reading, not compiled, not run on hardware
//
#include <hip/hip_runtime.h>
#include <hip/hip_bf16.h>

// ---------------------------------------------------------------------------
// CenterHead (2x conv3x3 128->128 + BN + ReLU, hm/reg heads) via bf16 WMMA
// implicit GEMM with LDS tile staging and 2Mx2N register tiling,
// plus target assignment.  gfx1250: wave32, v_wmma_f32_16x16x32_bf16.
// ---------------------------------------------------------------------------

typedef __attribute__((ext_vector_type(16))) __bf16 v16bf;
typedef __attribute__((ext_vector_type(8)))  __bf16 v8bf;
typedef __attribute__((ext_vector_type(8)))  float  v8f;

#define FW_   180
#define FH_   180
#define CIN_  128
#define BATCH_ 4
#define M_    500
#define PH_   182          // padded rows (1 halo each side)
#define PW_   194          // padded cols (halo + x-tile overshoot room)
#define XT_   12           // 16-pixel x tiles per row (head kernel)
#define XT64_ 3            // 64-pixel x tiles per row (conv kernel)
#define OV_   0.1f

// LDS staging tile: 3 rows x 66 cols x 128 ch (bf16) = 50,688 B
#define TILE_COLS 66
#define TILE_ELEMS (3 * TILE_COLS * CIN_)
#define TILE_VEC16 (TILE_ELEMS / 8)      // 16-byte vectors = 3168

// ---- d_out layout (floats, reference return order) ----
#define OFF_HM    ((size_t)0)
#define OFF_REG   ((size_t)259200)                 // 4*2*180*180
#define OFF_HEAT  (OFF_REG  + (size_t)1166400)     // 4*9*180*180
#define OFF_RB    (OFF_HEAT + (size_t)259200)
#define OFF_INDS  (OFF_RB   + (size_t)80000)       // 4*2500*8
#define OFF_MASK  (OFF_INDS + (size_t)10000)
#define OFF_HMASK (OFF_MASK + (size_t)10000)

// ---- workspace layout (bf16 elements first, then floats) ----
static constexpr size_t XPAD_ELEMS = (size_t)BATCH_ * PH_ * PW_ * CIN_; // 18,077,696
static constexpr size_t WP_ELEMS   = (size_t)8 * 9 * 4 * 32 * 16;       // 147,456
static constexpr size_t WHP_ELEMS  = (size_t)9 * 4 * 32 * 16;           // 18,432
static constexpr size_t OFF_XPAD = 0;
static constexpr size_t OFF_F1   = OFF_XPAD + XPAD_ELEMS;
static constexpr size_t OFF_F2   = OFF_F1 + XPAD_ELEMS;
static constexpr size_t OFF_W1   = OFF_F2 + XPAD_ELEMS;
static constexpr size_t OFF_W2   = OFF_W1 + WP_ELEMS;
static constexpr size_t OFF_WH   = OFF_W2 + WP_ELEMS;
static constexpr size_t BF16_TOTAL = OFF_WH + WHP_ELEMS;

// ---------------------------------------------------------------------------
// Pack input x (f32 NCHW) -> bf16 channels-last with zero halo.
// ---------------------------------------------------------------------------
__global__ __launch_bounds__(256)
void pack_input_kernel(const float* __restrict__ x, __bf16* __restrict__ xpad)
{
    size_t t = (size_t)blockIdx.x * blockDim.x + threadIdx.x;
    if (t >= XPAD_ELEMS) return;
    int c  = (int)(t % CIN_);
    int xx = (int)((t / CIN_) % PW_);
    int yy = (int)((t / ((size_t)CIN_ * PW_)) % PH_);
    int b  = (int)(t / ((size_t)CIN_ * PW_ * PH_));
    float v = 0.f;
    if (yy >= 1 && yy <= FH_ && xx >= 1 && xx <= FW_)
        v = x[(((size_t)b * CIN_ + c) * FH_ + (yy - 1)) * FW_ + (xx - 1)];
    xpad[t] = (__bf16)v;
}

__global__ __launch_bounds__(256)
void zero_bf16_kernel(__bf16* __restrict__ p, size_t n)
{
    size_t t = (size_t)blockIdx.x * blockDim.x + threadIdx.x;
    if (t < n) p[t] = (__bf16)0.f;
}

__global__ __launch_bounds__(256)
void zero_f32_kernel(float* __restrict__ p, int n)
{
    int t = blockIdx.x * blockDim.x + threadIdx.x;
    if (t < n) p[t] = 0.f;
}

// ---------------------------------------------------------------------------
// Pack conv weights (OIHW f32) into WMMA B-fragment lane layout, bf16.
// Chunk [(ot*9+tap)*4+icc]: lane l holds 16 K values:
//   N = l&15 (oc within tile),  K = (l>=16 ? 16 : 0) + j  (ic within chunk).
// ---------------------------------------------------------------------------
__global__ __launch_bounds__(256)
void pack_convw_kernel(const float* __restrict__ w, __bf16* __restrict__ wp)
{
    int t = blockIdx.x * blockDim.x + threadIdx.x;
    if (t >= (int)WP_ELEMS) return;
    int j    = t % 16;
    int lane = (t / 16) % 32;
    int icc  = (t / 512) % 4;
    int tap  = (t / 2048) % 9;
    int ot   = t / 18432;
    int oc = ot * 16 + (lane & 15);
    int ic = icc * 32 + (lane >> 4) * 16 + j;
    wp[t] = (__bf16)w[((size_t)oc * CIN_ + ic) * 9 + tap];
}

// Combined head weights: N 0..1 = hm, 2..10 = reg, 11..15 = zero.
__global__ __launch_bounds__(256)
void pack_headw_kernel(const float* __restrict__ hm_w,
                       const float* __restrict__ reg_w,
                       __bf16* __restrict__ wp)
{
    int t = blockIdx.x * blockDim.x + threadIdx.x;
    if (t >= (int)WHP_ELEMS) return;
    int j    = t % 16;
    int lane = (t / 16) % 32;
    int icc  = (t / 512) % 4;
    int tap  = t / 2048;
    int n  = lane & 15;
    int ic = icc * 32 + (lane >> 4) * 16 + j;
    float v = 0.f;
    if (n < 2)       v = hm_w[((size_t)n * CIN_ + ic) * 9 + tap];
    else if (n < 11) v = reg_w[((size_t)(n - 2) * CIN_ + ic) * 9 + tap];
    wp[t] = (__bf16)v;
}

// Fold BN into per-channel scale/shift; build head bias vector.
__global__ void bn_prep_kernel(const float* g1, const float* b1, const float* m1, const float* v1,
                               const float* g2, const float* b2, const float* m2, const float* v2,
                               const float* hm_b, const float* reg_b,
                               float* inv1, float* sh1, float* inv2, float* sh2, float* hsh)
{
    int c = threadIdx.x;
    if (c < CIN_) {
        float i1 = g1[c] * __frsqrt_rn(v1[c] + 1e-5f);
        inv1[c] = i1; sh1[c] = b1[c] - m1[c] * i1;
        float i2 = g2[c] * __frsqrt_rn(v2[c] + 1e-5f);
        inv2[c] = i2; sh2[c] = b2[c] - m2[c] * i2;
    }
    if (c < 16) hsh[c] = (c < 2) ? hm_b[c] : ((c < 11) ? reg_b[c - 2] : 0.f);
}

// ---------------------------------------------------------------------------
// Conv 128->128 (3x3, SAME) as implicit GEMM, LDS staging + 2Mx2N tiling.
// Workgroup = 8 waves = 64 output pixels (one row segment) x all 128 oc.
// Stage: 3 rows x 66 cols x 128 ch bf16 -> LDS (50.7 KB), once per WG.
// Wave (pg,og): pixels x0+32*pg..+31, oc 32*og..+31.  Per ic-chunk it loads
// A0/A1 (LDS) + B0/B1 (L2 weights) and issues 4 WMMAs -> 144 WMMAs/wave,
// 1 ds_load_b128 + 1 global_load_b128 per WMMA.
// Epilogue: BN scale/shift + ReLU -> bf16 channels-last.
// ---------------------------------------------------------------------------
__global__ __launch_bounds__(256)
void conv_wmma_kernel(const __bf16* __restrict__ src,
                      const __bf16* __restrict__ wpk,
                      const float*  __restrict__ inv,
                      const float*  __restrict__ sh,
                      __bf16* __restrict__ dst)
{
    __shared__ __bf16 smem[TILE_ELEMS];

    int wg = blockIdx.x;
    int xt = wg % XT64_;
    int y  = (wg / XT64_) % FH_;
    int b  = wg / (XT64_ * FH_);
    int x0 = xt * 64;
    int tid  = threadIdx.x;
    int lane = tid & 31;
    int wave = tid >> 5;
    int pg = wave & 1;                     // pixel group (0: +0..31, 1: +32..63)
    int og = wave >> 1;                    // oc group (32 oc each)
    int m  = lane & 15;                    // pixel row of A held by this lane
    int hi = lane >> 4;

    // ---- cooperative stage: padded rows y..y+2, padded cols x0..x0+65 ----
    {
        const uint4* gsrc = (const uint4*)src;
        uint4* sdst = (uint4*)smem;
        size_t gbase = (((size_t)b * PH_ + y) * PW_ + x0) * (CIN_ / 8);
        #pragma unroll
        for (int it = 0; it < 13; ++it) {
            int v = tid + it * 256;
            if (v < TILE_VEC16) {
                int r  = v / (TILE_COLS * (CIN_ / 8));
                int rm = v % (TILE_COLS * (CIN_ / 8));
                sdst[v] = gsrc[gbase + (size_t)r * PW_ * (CIN_ / 8) + rm];
            }
        }
    }
    __syncthreads();

    v8f acc00, acc01, acc10, acc11;        // [M-tile][N-tile]
    #pragma unroll
    for (int i = 0; i < 8; ++i) {
        acc00[i] = 0.f; acc01[i] = 0.f; acc10[i] = 0.f; acc11[i] = 0.f;
    }

    const __bf16* wp0 = wpk + (size_t)(2 * og)     * 9 * 4 * 32 * 16;
    const __bf16* wp1 = wpk + (size_t)(2 * og + 1) * 9 * 4 * 32 * 16;

    #pragma unroll
    for (int tap = 0; tap < 9; ++tap) {
        int ky = tap / 3, kx = tap % 3;
        const __bf16* row = smem + (size_t)(ky * TILE_COLS) * CIN_;
        const __bf16* a0p = row + (size_t)(pg * 32 + m + kx) * CIN_;
        const __bf16* a1p = a0p + (size_t)16 * CIN_;
        #pragma unroll
        for (int icc = 0; icc < 4; ++icc) {
            int ic0 = icc * 32 + hi * 16;  // contiguous K: k = 16*hi + s (matches packed B)
            v8bf a0lo = *(const v8bf*)(a0p + ic0);
            v8bf a0hi = *(const v8bf*)(a0p + ic0 + 8);
            v8bf a1lo = *(const v8bf*)(a1p + ic0);
            v8bf a1hi = *(const v8bf*)(a1p + ic0 + 8);
            v16bf a0 = __builtin_shufflevector(a0lo, a0hi,
                0,1,2,3,4,5,6,7,8,9,10,11,12,13,14,15);
            v16bf a1 = __builtin_shufflevector(a1lo, a1hi,
                0,1,2,3,4,5,6,7,8,9,10,11,12,13,14,15);
            size_t woff = ((size_t)(tap * 4 + icc) * 32 + lane) * 16;
            v16bf b0 = *(const v16bf*)(wp0 + woff);
            v16bf b1 = *(const v16bf*)(wp1 + woff);
            acc00 = __builtin_amdgcn_wmma_f32_16x16x32_bf16(
                false, a0, false, b0, (short)0, acc00, false, false);
            acc01 = __builtin_amdgcn_wmma_f32_16x16x32_bf16(
                false, a0, false, b1, (short)0, acc01, false, false);
            acc10 = __builtin_amdgcn_wmma_f32_16x16x32_bf16(
                false, a1, false, b0, (short)0, acc10, false, false);
            acc11 = __builtin_amdgcn_wmma_f32_16x16x32_bf16(
                false, a1, false, b1, (short)0, acc11, false, false);
        }
    }

    int n   = lane & 15;
    int oc0 = og * 32 + n;
    int oc1 = oc0 + 16;
    float iv0 = inv[oc0], sv0 = sh[oc0];
    float iv1 = inv[oc1], sv1 = sh[oc1];
    int mbase = hi * 8;                    // C/D layout: lanes 16-31 -> M=8..15
    size_t drow = ((size_t)b * PH_ + (y + 1)) * PW_;
    int pxb = x0 + pg * 32;
    #pragma unroll
    for (int r = 0; r < 8; ++r) {
        int px0 = pxb + mbase + r;
        if (px0 < FW_) {
            size_t o = (drow + (px0 + 1)) * CIN_;
            dst[o + oc0] = (__bf16)fmaxf(acc00[r] * iv0 + sv0, 0.f);
            dst[o + oc1] = (__bf16)fmaxf(acc01[r] * iv1 + sv1, 0.f);
        }
        int px1 = pxb + 16 + mbase + r;
        if (px1 < FW_) {
            size_t o = (drow + (px1 + 1)) * CIN_;
            dst[o + oc0] = (__bf16)fmaxf(acc10[r] * iv0 + sv0, 0.f);
            dst[o + oc1] = (__bf16)fmaxf(acc11[r] * iv1 + sv1, 0.f);
        }
    }
}

// ---------------------------------------------------------------------------
// Head conv: 128 -> 16 (2 hm + 9 reg + 5 zero) in one WMMA N-tile per wave.
// ---------------------------------------------------------------------------
__global__ __launch_bounds__(256)
void head_wmma_kernel(const __bf16* __restrict__ src,
                      const __bf16* __restrict__ whp,
                      const float*  __restrict__ hsh,
                      float* __restrict__ out)
{
    int gid  = blockIdx.x * 8 + (threadIdx.x >> 5);   // tile id (8640 total)
    int lane = threadIdx.x & 31;
    int xt = gid % XT_;
    int y  = (gid / XT_) % FH_;
    int b  = gid / (XT_ * FH_);
    int x0 = xt * 16;
    int m  = lane & 15;
    int hi = lane >> 4;

    v8f acc;
    #pragma unroll
    for (int i = 0; i < 8; ++i) acc[i] = 0.f;

    #pragma unroll
    for (int tap = 0; tap < 9; ++tap) {
        int ky = tap / 3, kx = tap % 3;
        const __bf16* rowp =
            src + (((size_t)b * PH_ + (y + ky)) * PW_ + (x0 + m + kx)) * CIN_;
        #pragma unroll
        for (int icc = 0; icc < 4; ++icc) {
            int ic0 = icc * 32 + hi * 16;
            v8bf lo = *(const v8bf*)(rowp + ic0);
            v8bf hj = *(const v8bf*)(rowp + ic0 + 8);
            v16bf a = __builtin_shufflevector(lo, hj,
                0,1,2,3,4,5,6,7,8,9,10,11,12,13,14,15);
            v16bf bf = *(const v16bf*)(whp + ((size_t)(tap * 4 + icc) * 32 + lane) * 16);
            acc = __builtin_amdgcn_wmma_f32_16x16x32_bf16(
                false, a, false, bf, (short)0, acc, false, false);
        }
    }

    int n = lane & 15;
    if (n >= 11) return;
    float bias = hsh[n];
    int mbase = hi * 8;
    #pragma unroll
    for (int r = 0; r < 8; ++r) {
        int px = x0 + mbase + r;
        if (px < FW_) {
            float v = acc[r] + bias;
            if (n < 2)
                out[OFF_HM  + ((size_t)(b * 2 + n)     * FH_ + y) * FW_ + px] = v;
            else
                out[OFF_REG + ((size_t)(b * 9 + n - 2) * FH_ + y) * FW_ + px] = v;
        }
    }
}

// ---------------------------------------------------------------------------
// Target assignment: per-object scalars, ret_boxes/inds/mask (5 slots).
// ---------------------------------------------------------------------------
__global__ void targets_kernel(const float* __restrict__ gt,
                               float* __restrict__ out,
                               float* __restrict__ objp)
{
    int o = blockIdx.x * blockDim.x + threadIdx.x;
    if (o >= BATCH_ * M_) return;
    int b = o / M_, mi = o % M_;
    const float* g = gt + (size_t)o * 8;
    float x = g[0], y = g[1], z = g[2];
    float dx = g[3], dy = g[4], dz = g[5], ry = g[6];
    int cls = (int)g[7];

    float coord_x = fminf(fmaxf((x + 54.f) / 0.075f / 8.f, 0.f), 179.5f);
    float coord_y = fminf(fmaxf((y + 54.f) / 0.075f / 8.f, 0.f), 179.5f);
    int cxi = (int)coord_x, cyi = (int)coord_y;
    float h = dx / 0.6f, w = dy / 0.6f;   // dxp, dyp

    // gaussian_radius(height=dxp, width=dyp, OVERLAP)
    float b1 = h + w, c1 = w * h * (1.f - OV_) / (1.f + OV_);
    float r1 = (b1 + sqrtf(fmaxf(b1 * b1 - 4.f * c1, 0.f))) * 0.5f;
    float b2 = 2.f * (h + w), c2 = (1.f - OV_) * w * h;
    float r2 = (b2 + sqrtf(fmaxf(b2 * b2 - 16.f * c2, 0.f))) * 0.5f;
    float a3 = 4.f * OV_, b3 = -2.f * OV_ * (h + w), c3 = (OV_ - 1.f) * w * h;
    float r3 = (b3 + sqrtf(fmaxf(b3 * b3 - 4.f * a3 * c3, 0.f))) / (2.f * a3);
    float rf = fminf(fminf(r1, r2), r3);
    int radius = (int)rf; if (radius < 2) radius = 2;

    bool valid = (h > 0.f) && (w > 0.f);
    bool pos = valid && (cls >= 1);

    int base_ind = cyi * FW_ + cxi;
    bool w_cond = (0.075f * 8.f) > dy * 0.25f;
    bool l_cond = (0.075f * 8.f) > dx * 0.25f;
    bool left = (cxi - 1) >= 0, right = (cxi + 1) < FW_;
    bool up   = (cyi - 1) >= 0, down  = (cyi + 1) < FH_;
    int m0 = valid ? (cls >= 1 ? 1 : -1) : 0;
    bool s1 = valid && !w_cond && left;
    bool s2 = valid && !w_cond && right;
    bool s3 = valid && !l_cond && up;
    bool s4 = valid && !l_cond && down;

    float offx = coord_x - (float)cxi, offy = coord_y - (float)cyi;
    float base8[8] = { offx, offy, z,
                       logf(fmaxf(dx, 1e-8f)), logf(fmaxf(dy, 1e-8f)),
                       logf(fmaxf(dz, 1e-8f)), cosf(ry), sinf(ry) };
    int   maskv[5] = { m0, s1 ? m0 : 0, s2 ? m0 : 0, s3 ? m0 : 0, s4 ? m0 : 0 };
    int   indv[5]  = { valid ? base_ind : 0,
                       s1 ? base_ind - 1   : 0, s2 ? base_ind + 1   : 0,
                       s3 ? base_ind - FW_ : 0, s4 ? base_ind + FW_ : 0 };
    float dxa[5] = { 0.f, s1 ? 1.f : 0.f, s2 ? -1.f : 0.f, 0.f, 0.f };
    float dya[5] = { 0.f, 0.f, 0.f, s3 ? 1.f : 0.f, s4 ? -1.f : 0.f };
    float vf = valid ? 1.f : 0.f;

    size_t slot = (size_t)b * (5 * M_) + (size_t)mi * 5;
    for (int s = 0; s < 5; ++s) {
        out[OFF_MASK + slot + s] = (float)maskv[s];
        out[OFF_INDS + slot + s] = (float)indv[s];
        float* rb = out + OFF_RB + (slot + s) * 8;
        rb[0] = (base8[0] + dxa[s]) * vf;
        rb[1] = (base8[1] + dya[s]) * vf;
        #pragma unroll
        for (int j = 2; j < 8; ++j) rb[j] = base8[j] * vf;
    }

    float* p = objp + (size_t)o * 8;
    p[0] = pos ? 1.f : 0.f;
    p[1] = (float)(cls - 1);
    p[2] = (float)cxi; p[3] = (float)cyi; p[4] = (float)radius;
}

// Gaussian splat, max-reduced per class: one wave per object, atomicMax on
// int bit pattern (valid for non-negative floats; deterministic).
__global__ __launch_bounds__(256)
void splat_kernel(const float* __restrict__ objp, float* __restrict__ out)
{
    int o = blockIdx.x * 8 + (threadIdx.x >> 5);
    if (o >= BATCH_ * M_) return;
    const float* p = objp + (size_t)o * 8;
    if (p[0] == 0.f) return;
    int b = o / M_;
    int cidx = (int)p[1], cxi = (int)p[2], cyi = (int)p[3], r = (int)p[4];
    float sigma = (2.f * (float)r + 1.f) / 6.f;
    float k = 1.f / (2.f * sigma * sigma);
    int wlen = 2 * r + 1, cells = wlen * wlen;
    int lane = threadIdx.x & 31;
    int* heat = (int*)(out + OFF_HEAT);
    for (int i = lane; i < cells; i += 32) {
        int ddy = i / wlen - r, ddx = i % wlen - r;
        int yy = cyi + ddy, xx = cxi + ddx;
        if (yy >= 0 && yy < FH_ && xx >= 0 && xx < FW_) {
            float gv = __expf(-(float)(ddx * ddx + ddy * ddy) * k);
            atomicMax(heat + ((size_t)(b * 2 + cidx) * FH_ + yy) * FW_ + xx,
                      __float_as_int(gv));
        }
    }
}

__global__ __launch_bounds__(256)
void hmask_kernel(float* __restrict__ out)
{
    int t = blockIdx.x * blockDim.x + threadIdx.x;
    if (t < 259200)
        out[OFF_HMASK + t] = (out[OFF_HEAT + t] > 0.f) ? 1.f : 0.f;
}

// ---------------------------------------------------------------------------
extern "C" void kernel_launch(void* const* d_in, const int* in_sizes, int n_in,
                              void* d_out, int out_size, void* d_ws, size_t ws_size,
                              hipStream_t stream)
{
    (void)in_sizes; (void)n_in; (void)out_size; (void)ws_size;
    const float* x     = (const float*)d_in[0];
    const float* gt    = (const float*)d_in[1];
    const float* c1w   = (const float*)d_in[2];
    const float* g1    = (const float*)d_in[3];
    const float* bb1   = (const float*)d_in[4];
    const float* mm1   = (const float*)d_in[5];
    const float* vv1   = (const float*)d_in[6];
    const float* c2w   = (const float*)d_in[7];
    const float* g2    = (const float*)d_in[8];
    const float* bb2   = (const float*)d_in[9];
    const float* mm2   = (const float*)d_in[10];
    const float* vv2   = (const float*)d_in[11];
    const float* hm_w  = (const float*)d_in[12];
    const float* hm_b  = (const float*)d_in[13];
    const float* reg_w = (const float*)d_in[14];
    const float* reg_b = (const float*)d_in[15];
    float* out = (float*)d_out;

    __bf16* bfb  = (__bf16*)d_ws;
    __bf16* xpad = bfb + OFF_XPAD;
    __bf16* f1   = bfb + OFF_F1;
    __bf16* f2   = bfb + OFF_F2;
    __bf16* w1   = bfb + OFF_W1;
    __bf16* w2   = bfb + OFF_W2;
    __bf16* wh   = bfb + OFF_WH;
    size_t fb = (BF16_TOTAL * 2 + 255) & ~(size_t)255;
    float* inv1 = (float*)((char*)d_ws + fb);
    float* sh1  = inv1 + 128;
    float* inv2 = sh1 + 128;
    float* sh2  = inv2 + 128;
    float* hsh  = sh2 + 128;
    float* objp = hsh + 16;     // BATCH_*M_*8 floats

    // --- prep ---
    bn_prep_kernel<<<1, 256, 0, stream>>>(g1, bb1, mm1, vv1, g2, bb2, mm2, vv2,
                                          hm_b, reg_b, inv1, sh1, inv2, sh2, hsh);
    pack_convw_kernel<<<(int)((WP_ELEMS + 255) / 256), 256, 0, stream>>>(c1w, w1);
    pack_convw_kernel<<<(int)((WP_ELEMS + 255) / 256), 256, 0, stream>>>(c2w, w2);
    pack_headw_kernel<<<(int)((WHP_ELEMS + 255) / 256), 256, 0, stream>>>(hm_w, reg_w, wh);
    pack_input_kernel<<<(int)((XPAD_ELEMS + 255) / 256), 256, 0, stream>>>(x, xpad);
    zero_bf16_kernel<<<(int)((XPAD_ELEMS + 255) / 256), 256, 0, stream>>>(f1, XPAD_ELEMS);
    zero_bf16_kernel<<<(int)((XPAD_ELEMS + 255) / 256), 256, 0, stream>>>(f2, XPAD_ELEMS);

    // --- conv tower (WMMA + LDS staging, 2Mx2N register tiles) ---
    int conv_grid = BATCH_ * FH_ * XT64_;             // 2160 workgroups
    conv_wmma_kernel<<<conv_grid, 256, 0, stream>>>(xpad, w1, inv1, sh1, f1);
    conv_wmma_kernel<<<conv_grid, 256, 0, stream>>>(f1,   w2, inv2, sh2, f2);
    head_wmma_kernel<<<BATCH_ * FH_ * XT_ / 8, 256, 0, stream>>>(f2, wh, hsh, out);

    // --- target assignment ---
    zero_f32_kernel<<<(259200 + 255) / 256, 256, 0, stream>>>(out + OFF_HEAT, 259200);
    targets_kernel<<<(BATCH_ * M_ + 255) / 256, 256, 0, stream>>>(gt, out, objp);
    splat_kernel<<<(BATCH_ * M_ + 7) / 8, 256, 0, stream>>>(objp, out);
    hmask_kernel<<<(259200 + 255) / 256, 256, 0, stream>>>(out);
}
